// UniPhyFluidScan_781684048025
// MI455X (gfx1250) — matrix operand order repaired and checked by reading, not compiled, not run on hardware
//
#include <hip/hip_runtime.h>
#include <hip/hip_bf16.h>
#include <cstdint>

// ---------------------------------------------------------------------------
// UniPhyFluidScan on MI455X (gfx1250, wave32, WMMA).
// phase-MLP -> 4 WMMA convs (analytic backward of interaction) ->
// rfft2 as WMMA GEMMs -> constrain -> scan -> constrain -> norm ->
// irfft2 as WMMA GEMMs.  f16 operands / f32 accumulate.
// All fragment global loads are unconditional dword loads (guards replaced by
// pointer clamping / pre-zeroed padding) to avoid branchy u16-load codegen.
// ---------------------------------------------------------------------------

typedef _Float16 half_t;
typedef __attribute__((ext_vector_type(16))) _Float16 v16h;
typedef __attribute__((ext_vector_type(8)))  float    v8f;
typedef __attribute__((ext_vector_type(8)))  uint32_t v8u;
typedef __attribute__((ext_vector_type(4)))  uint32_t v4u;

#define BB 2
#define TT 32
#define CC 8
#define HH 192
#define WW 192
#define WF 97
#define HW  (HH*WW)       // 36864
#define HWF (HH*WF)       // 18624
#define NIMG (BB*TT)      // 64
#define NSLICE (NIMG*CC)  // 512

__device__ __forceinline__ float sigmoidf_(float x){ return 1.f/(1.f+__expf(-x)); }
__device__ __forceinline__ float siluf_(float x){ return x*sigmoidf_(x); }
__device__ __forceinline__ float dsiluf_(float x){ float s=sigmoidf_(x); return s*(1.f+x*(1.f-s)); }

__device__ __forceinline__ v8f wmma16(v16h a, v16h b, v8f c){
  return __builtin_amdgcn_wmma_f32_16x16x32_f16(false, a, false, b, (short)0, c, false, false);
}
// CDNA5 ISA 7.12.2 fragment K-index helpers (wave32); both return even K.
__device__ __forceinline__ int a_kbase(int v, int lane){
  return ((v<4)? 2*v : 16 + 2*(v-4)) + ((lane>=16)?8:0);
}
__device__ __forceinline__ int b_kbase(int v, int lane){
  return ((v<4)? 2*v : 16 + 2*(v-4)) + ((lane>=16)?8:0);
}
__device__ __forceinline__ uint32_t ldg32(const half_t* p){ return *(const uint32_t*)p; }
__device__ __forceinline__ v16h as_h(v8u u){ return __builtin_bit_cast(v16h, u); }
__device__ __forceinline__ uint32_t pk2(float a, float b){
  uint16_t lo = __builtin_bit_cast(uint16_t, (half_t)a);
  uint16_t hi = __builtin_bit_cast(uint16_t, (half_t)b);
  return (uint32_t)lo | ((uint32_t)hi<<16);
}

// ---------------------------------------------------------------------------
// Twiddle matrices (f16), ortho scaling baked in.
// ---------------------------------------------------------------------------
__global__ void k_twid(half_t* __restrict__ Mw1t, half_t* __restrict__ Fr,
                       half_t* __restrict__ Fi,   half_t* __restrict__ Gr,
                       half_t* __restrict__ Gi,   half_t* __restrict__ M2t)
{
  int gid = blockIdx.x*blockDim.x + threadIdx.x;
  const float S  = 0.07216878364870323f;       // 1/sqrt(192)
  const float W2 = 6.283185307179586f/192.f;
  const int R0 = 208*192;
  const int R1 = R0 + 192*192;
  const int R2 = R1 + 192*192;
  const int R3 = R2 + 192*224;
  if (gid < R0){
    int n = gid/192, w = gid%192;
    float v = 0.f;
    if (n < 194){
      int kx = n>>1; float th = W2*(float)((w*kx)%192);
      v = ((n&1) ? -sinf(th) : cosf(th)) * S;
    }
    Mw1t[gid] = (half_t)v;
  } else if (gid < R1){
    int id = gid - R0; int ky = id/192, h = id%192;
    float th = W2*(float)((h*ky)%192);
    Fr[id] = (half_t)( cosf(th)*S);
    Fi[id] = (half_t)(-sinf(th)*S);
  } else if (gid < R2){
    int id = gid - R1; int h = id/192, ky = id%192;
    float th = W2*(float)((h*ky)%192);
    Gr[id] = (half_t)( cosf(th)*S);
    Gi[id] = (half_t)( sinf(th)*S);
  } else if (gid < R3){
    int id = gid - R2; int w = id/224, k = id%224;
    float v = 0.f;
    if (k < 194){
      int kx = k>>1;
      float al = (kx==0 || kx==96) ? 1.f : 2.f;  // Hermitian fold
      float th = W2*(float)((w*kx)%192);
      v = ((k&1) ? -sinf(th) : cosf(th)) * al * S;
    }
    M2t[id] = (half_t)v;
  }
}

// Zero X1 K-pad tail [194,224) so inverse-row GEMM needs no K guards.
__global__ void k_padX1(half_t* __restrict__ X1)
{
  int gid = blockIdx.x*blockDim.x + threadIdx.x;
  const int PAD = 30;
  if (gid >= NSLICE*HH*PAD) return;
  int row = gid / PAD, k = 194 + gid % PAD;
  X1[(size_t)row*224 + k] = (half_t)0.f;
}

// ---------------------------------------------------------------------------
// Phase MLP -> complex decay A[b][c][h*WF+j].
// ---------------------------------------------------------------------------
__global__ void k_phase(const float* __restrict__ dtp, const float* __restrict__ w1,
                        const float* __restrict__ b1,  const float* __restrict__ w2,
                        const float* __restrict__ b2,  float2* __restrict__ Aop)
{
  int gid = blockIdx.x*blockDim.x + threadIdx.x;
  if (gid >= HWF) return;
  int h = gid / WF, j = gid % WF;
  float ky = (float)((h < 96) ? h : (h - 192)) * (1.f/192.f);
  float kx = (float)j * (1.f/192.f);
  float acc[8];
  #pragma unroll
  for (int c=0;c<8;++c) acc[c]=0.f;
  for (int u=0;u<64;++u){
    float su = siluf_(ky*w1[u] + kx*w1[64+u] + b1[u]);
    #pragma unroll
    for (int c=0;c<8;++c) acc[c] += su * w2[u*8 + c];
  }
  float kp = sqrtf(ky*ky + kx*kx);
  #pragma unroll
  for (int c=0;c<8;++c){
    float om = acc[c] + b2[c] + kp;
    #pragma unroll
    for (int b=0;b<2;++b){
      float ph = om * dtp[b];
      Aop[(size_t)(b*8 + c)*HWF + gid] = make_float2(cosf(ph), sinf(ph));
    }
  }
}

// ---------------------------------------------------------------------------
// 3x3 conv as WMMA GEMM: M=16 out-ch, N=16 pixels, K=Cin*9 (padded).
// Patch has CINP >= KPAD/9 channels; channels >= CIN are zero so im2col
// gathers need no K guard.
// ---------------------------------------------------------------------------
template<int CIN,int COUT,int KREAL,int KPAD,int WDIV,int LOADMODE,int WMODE,int POST,int HASBIAS>
__global__ void __launch_bounds__(128)
k_conv(const void* __restrict__ src, const float* __restrict__ wsrc,
       const float* __restrict__ bias, half_t* __restrict__ out,
       const half_t* __restrict__ auxZ, const float* __restrict__ c3w,
       const float* __restrict__ xin, const float* __restrict__ dtp)
{
  constexpr int CINP = (KPAD + 8) / 9;     // zero-padded channel count
  __shared__ half_t patch[CINP][18][20];
  __shared__ half_t wlds[16][KPAD];
  __shared__ float  gvs[16];
  const int tid  = threadIdx.x;
  const int img  = blockIdx.y;
  const int tile = blockIdx.x;
  const int ty = tile / 12, tx = tile % 12;
  const int y0 = ty*16, x0 = tx*16;

  if (POST==1 && tid < 16){
    float s=0.f;
    #pragma unroll
    for (int o=0;o<8;++o) s += c3w[o*16 + tid];
    gvs[tid]=s;
  }
  if (tid==0) __builtin_prefetch(wsrc, 0, 0);
  for (int i=tid; i<16*KPAD; i+=128){
    int oc = i / KPAD, k = i % KPAD;
    float v = 0.f;
    if (k < KREAL && oc < COUT){
      int ic = k/9, tap = k%9;
      int idx = (WMODE==0) ? ((oc*CIN + ic)*9 + tap)
                           : ((ic*WDIV + oc)*9 + (8 - tap));
      v = wsrc[idx];
    }
    wlds[oc][k] = (half_t)v;
  }
  for (int i=tid; i<CINP*18*18; i+=128){
    int ic = i/324, r = i%324, yy=r/18, xx=r%18;
    int gy = y0-1+yy, gx = x0-1+xx;
    float v = 0.f;
    if (ic < CIN && gy>=0 && gy<HH && gx>=0 && gx<WW){
      size_t gi = ((size_t)(img*CIN+ic)*HH + gy)*WW + gx;
      if (LOADMODE==0) v = ((const float*)src)[gi];
      else             v = (float)((const half_t*)src)[gi];
      if (LOADMODE==2) v = siluf_(v);
    }
    patch[ic][yy][xx] = (half_t)v;
  }
  __syncthreads();

  const int lane = tid & 31, wave = tid >> 5;
  v16h afrag[KPAD/32];
  {
    const int m = lane & 15;
    #pragma unroll
    for (int kc=0;kc<KPAD/32;++kc){
      v8u au;
      #pragma unroll
      for (int v=0;v<8;++v)
        au[v] = *(const uint32_t*)&wlds[m][kc*32 + a_kbase(v, lane)];
      afrag[kc] = as_h(au);
    }
  }
  #pragma unroll
  for (int row=0; row<4; ++row){
    const int py = wave*4 + row;
    const int px = lane & 15;
    v8f acc = {};
    #pragma unroll
    for (int kc=0;kc<KPAD/32;++kc){
      v16h bfrag;
      #pragma unroll
      for (int v=0;v<8;++v){
        int kb = kc*32 + b_kbase(v, lane);
        #pragma unroll
        for (int e=0;e<2;++e){
          int k = kb + e;
          int ic = k/9, tap = k%9;
          bfrag[2*v+e] = patch[ic][py + tap/3][px + tap%3];
        }
      }
      acc = wmma16(afrag[kc], bfrag, acc);
    }
    const int gy = y0 + py, gx = x0 + px;
    #pragma unroll
    for (int r=0;r<8;++r){
      int oc = r + ((lane>=16)?8:0);
      if (COUT == 16 || oc < COUT){
        float z = acc[r] + (HASBIAS ? bias[oc] : 0.f);
        size_t gi = ((size_t)(img*COUT+oc)*HH + gy)*WW + gx;
        float o;
        if (POST==0)      o = z;
        else if (POST==1) o = gvs[oc] * dsiluf_(z);
        else if (POST==2) o = z * dsiluf_((float)auxZ[gi]);
        else              o = xin[gi] + z * dtp[img / TT];
        out[gi] = (half_t)o;
      }
    }
  }
}

// ---------------------------------------------------------------------------
// Forward row rfft: S1[slice][ri][kx][h] = sum_w y[h][w]*Mw1t[2kx+ri][w].
// B row pointer clamped (OOB columns never stored). Packed b128 store.
// ---------------------------------------------------------------------------
__global__ void __launch_bounds__(128)
k_fft_rows(const half_t* __restrict__ yf, const half_t* __restrict__ Bw,
           half_t* __restrict__ S1)
{
  const int slice = blockIdx.z;
  const int mt    = blockIdx.y;
  const int lane  = threadIdx.x & 31, wave = threadIdx.x >> 5;
  const int ntile = blockIdx.x*4 + wave;
  if (ntile >= 13) return;
  const int m  = lane & 15;
  const half_t* ap = yf + (size_t)slice*HW + (size_t)(mt*16 + m)*WW;
  const int nn = ntile*16 + (lane & 15);
  const bool nok = (nn < 194);
  const half_t* bp = Bw + (size_t)(nok ? nn : 193)*WW;   // clamped, branch-free
  __builtin_prefetch(ap, 0, 0);
  v8f acc = {};
  #pragma unroll
  for (int kc=0;kc<6;++kc){
    v8u au, bu;
    #pragma unroll
    for (int v=0;v<8;++v){
      au[v] = ldg32(ap + kc*32 + a_kbase(v,lane));
      bu[v] = ldg32(bp + kc*32 + b_kbase(v,lane));
    }
    acc = wmma16(as_h(au), as_h(bu), acc);
  }
  if (nok){
    int ri = nn & 1, kx = nn >> 1;
    half_t* op = S1 + (size_t)slice*2*HWF + (size_t)ri*HWF + (size_t)kx*HH
               + mt*16 + ((lane>=16)?8:0);
    v4u pkv;
    #pragma unroll
    for (int r=0;r<4;++r) pkv[r] = pk2(acc[2*r], acc[2*r+1]);
    *(v4u*)op = pkv;                                      // b128 store
  }
}

// ---------------------------------------------------------------------------
// Complex GEMM along H: out = (Ar + iAi) @ (B0 + iB1).
// OUTFMT 0: float2 Xf[slice][row][kx];  1: f16 interleaved X1 (stride 224).
// ---------------------------------------------------------------------------
template<int OUTFMT>
__global__ void __launch_bounds__(128)
k_cgemm(const half_t* __restrict__ Ar, const half_t* __restrict__ Ai,
        const half_t* __restrict__ Bb, void* __restrict__ outv)
{
  const int slice = blockIdx.z;
  const int mt    = blockIdx.y;
  const int lane  = threadIdx.x & 31, wave = threadIdx.x >> 5;
  const int ntile = blockIdx.x*4 + wave;
  if (ntile >= 7) return;
  const half_t* B0 = Bb + (size_t)slice*2*HWF;
  const half_t* B1 = B0 + HWF;
  const int m  = lane & 15;
  const half_t* arp = Ar + (size_t)(mt*16+m)*HH;
  const half_t* aip = Ai + (size_t)(mt*16+m)*HH;
  const int nn = ntile*16 + (lane&15);
  const bool nok = nn < WF;
  const int nns = nok ? nn : (WF-1);                     // clamped row
  const half_t* b0p = B0 + (size_t)nns*HH;
  const half_t* b1p = B1 + (size_t)nns*HH;
  v8f aR = {}, aI = {};
  #pragma unroll
  for (int kc=0;kc<6;++kc){
    v8u aru, aiu, bru, biu;
    #pragma unroll
    for (int v=0;v<8;++v){
      int ka = kc*32 + a_kbase(v,lane);
      int kb = kc*32 + b_kbase(v,lane);
      aru[v] = ldg32(arp + ka);
      aiu[v] = ldg32(aip + ka);
      bru[v] = ldg32(b0p + kb);
      biu[v] = ldg32(b1p + kb);
    }
    v8u ainu = aiu ^ 0x80008000u;                        // conj via sign xor
    aR = wmma16(as_h(aru),  as_h(bru), aR);
    aR = wmma16(as_h(ainu), as_h(biu), aR);              // Re += -Ai*Bi
    aI = wmma16(as_h(aru),  as_h(biu), aI);
    aI = wmma16(as_h(aiu),  as_h(bru), aI);
  }
  if (nok){
    #pragma unroll
    for (int r=0;r<8;++r){
      int rrow = mt*16 + r + ((lane>=16)?8:0);
      if (OUTFMT==0){
        float2* o = (float2*)outv + (size_t)slice*HWF + (size_t)rrow*WF + nn;
        *o = make_float2(aR[r], aI[r]);
      } else {
        half_t* o = (half_t*)outv + ((size_t)slice*HH + rrow)*224 + 2*nn;
        *(uint32_t*)o = pk2(aR[r], aI[r]);               // b32 packed store
      }
    }
  }
}

// ---------------------------------------------------------------------------
// Inverse row stage (irfft along W). K-pads are pre-zeroed => no guards.
// ---------------------------------------------------------------------------
__global__ void __launch_bounds__(128)
k_ifft_rows(const half_t* __restrict__ X1, const half_t* __restrict__ M2t,
            float* __restrict__ outp)
{
  const int slice = blockIdx.z;
  const int mt    = blockIdx.y;
  const int lane  = threadIdx.x & 31, wave = threadIdx.x >> 5;
  const int ntile = blockIdx.x*4 + wave;
  if (ntile >= 12) return;
  const int m  = lane & 15;
  const half_t* ap = X1 + ((size_t)slice*HH + (mt*16+m))*224;
  const int nn = ntile*16 + (lane&15);
  const half_t* bp = M2t + (size_t)nn*224;
  __builtin_prefetch(ap, 0, 0);
  v8f acc = {};
  #pragma unroll
  for (int kc=0;kc<7;++kc){
    v8u au, bu;
    #pragma unroll
    for (int v=0;v<8;++v){
      au[v] = ldg32(ap + kc*32 + a_kbase(v,lane));
      bu[v] = ldg32(bp + kc*32 + b_kbase(v,lane));
    }
    acc = wmma16(as_h(au), as_h(bu), acc);
  }
  #pragma unroll
  for (int r=0;r<8;++r){
    int rrow = mt*16 + r + ((lane>=16)?8:0);
    outp[(size_t)slice*HW + (size_t)rrow*WW + nn] = acc[r];
  }
}

// ---------------------------------------------------------------------------
// Divergence projection on ch 0/1 + |.|^2 reduction into nacc[bt].
// ---------------------------------------------------------------------------
__global__ void k_constrain_norm(float2* __restrict__ buf, float* __restrict__ nacc)
{
  int bt = blockIdx.y;
  int p  = blockIdx.x*blockDim.x + threadIdx.x;
  float sum = 0.f;
  if (p < HWF){
    int h = p / WF, j = p % WF;
    float ky = (float)((h < 96) ? h : (h - 192)) * (1.f/192.f);
    float kx = (float)j * (1.f/192.f);
    float k2 = ky*ky + kx*kx;
    if (h==0 && j==0) k2 = 1.f;
    size_t base = (size_t)bt*8*HWF + p;
    float2 u = buf[base];
    float2 v = buf[base + HWF];
    float dr = (ky*u.x + kx*v.x)/k2;
    float di = (ky*u.y + kx*v.y)/k2;
    u.x -= ky*dr; u.y -= ky*di;
    v.x -= kx*dr; v.y -= kx*di;
    buf[base]       = u;
    buf[base + HWF] = v;
    sum = u.x*u.x + u.y*u.y + v.x*v.x + v.y*v.y;
    #pragma unroll
    for (int c=2;c<8;++c){
      float2 e = buf[base + (size_t)c*HWF];
      sum += e.x*e.x + e.y*e.y;
    }
  }
  for (int off=16; off>0; off>>=1) sum += __shfl_down(sum, off, 32);
  if ((threadIdx.x & 31)==0) atomicAdd(&nacc[bt], sum);
}

// h_t = A*h_{t-1} + X_t, A constant over t.
__global__ void k_scan(const float2* __restrict__ Aop, const float2* __restrict__ Xf,
                       float2* __restrict__ hf)
{
  int gid = blockIdx.x*blockDim.x + threadIdx.x;
  if (gid >= BB*CC*HWF) return;
  int b   = gid / (CC*HWF);
  int rem = gid % (CC*HWF);
  float2 a = Aop[gid];
  float hr = 0.f, hi = 0.f;
  for (int t=0;t<TT;++t){
    size_t idx = (size_t)(b*TT + t)*CC*HWF + rem;
    float2 x = Xf[idx];
    float nr = a.x*hr - a.y*hi + x.x;
    float ni = a.x*hi + a.y*hr + x.y;
    hr = nr; hi = ni;
    hf[idx] = make_float2(hr, hi);
  }
}

// Rescale and write transposed f16 planes Zt[slice][ri][kx][ky].
__global__ void k_finalize(const float2* __restrict__ hf, const float* __restrict__ nin,
                           const float* __restrict__ nout, half_t* __restrict__ Zt)
{
  int gid = blockIdx.x*blockDim.x + threadIdx.x;
  if (gid >= NSLICE*HWF) return;
  int slice = gid / HWF, p = gid % HWF;
  int bt = slice >> 3;
  float s = sqrtf(nin[bt]) / (sqrtf(nout[bt]) + 1e-6f);
  float2 v = hf[gid];
  int h = p / WF, j = p % WF;
  size_t o = (size_t)slice*2*HWF + (size_t)j*HH + h;
  Zt[o]       = (half_t)(v.x*s);
  Zt[o + HWF] = (half_t)(v.y*s);
}

__global__ void k_zero(float* __restrict__ p, int n){
  int i = blockIdx.x*blockDim.x + threadIdx.x;
  if (i < n) p[i] = 0.f;
}

// ---------------------------------------------------------------------------
extern "C" void kernel_launch(void* const* d_in, const int* in_sizes, int n_in,
                              void* d_out, int out_size, void* d_ws, size_t ws_size,
                              hipStream_t stream)
{
  const float* x   = (const float*)d_in[0];
  const float* dt  = (const float*)d_in[1];
  const float* w1  = (const float*)d_in[2];
  const float* b1  = (const float*)d_in[3];
  const float* w2  = (const float*)d_in[4];
  const float* b2  = (const float*)d_in[5];
  const float* c1w = (const float*)d_in[6];
  const float* c1b = (const float*)d_in[7];
  const float* c2w = (const float*)d_in[8];
  const float* c2b = (const float*)d_in[9];
  const float* c3w = (const float*)d_in[10];
  (void)in_sizes; (void)n_in; (void)out_size; (void)ws_size;

  char* ws = (char*)d_ws;
  size_t off = 0;
  auto take = [&](size_t bytes)->char* {
    char* p = ws + off;
    off = (off + bytes + 255) & ~(size_t)255;
    return p;
  };
  half_t* z1   = (half_t*)take(75497472ull);
  half_t* g2   = (half_t*)take(75497472ull);
  half_t* ga1  = (half_t*)take(75497472ull);
  half_t* yf   = (half_t*)take(38141952ull);
  half_t* S1   = (half_t*)take(38141952ull);
  float2* Xf   = (float2*)take(76283904ull);
  float2* Aop  = (float2*)take(2383872ull);
  half_t* Mw1t = (half_t*)take(79872ull);
  half_t* Fr   = (half_t*)take(73728ull);
  half_t* Fi   = (half_t*)take(73728ull);
  half_t* Gr   = (half_t*)take(73728ull);
  half_t* Gi   = (half_t*)take(73728ull);
  half_t* M2t  = (half_t*)take(86016ull);
  float*  nin  = (float*)take(512ull);
  float*  nout = nin + 64;
  float2* hf   = (float2*)z1;    // alias: z1+g2 dead after F3/F4
  half_t* X1   = (half_t*)ga1;   // alias: ga1 dead after F4
  half_t* Zt   = yf;             // alias: yf dead after row-fft

  k_twid <<<(156672+255)/256, 256, 0, stream>>>(Mw1t, Fr, Fi, Gr, Gi, M2t);
  k_phase<<<(HWF+127)/128,    128, 0, stream>>>(dt, w1, b1, w2, b2, Aop);

  dim3 cg(144, 64);
  k_conv<8,16,72,96,8,0,0,0,1><<<cg,128,0,stream>>>(x,  c1w, c1b, z1,  nullptr, nullptr, nullptr, nullptr);
  k_conv<16,16,144,160,16,2,0,1,1><<<cg,128,0,stream>>>(z1, c2w, c2b, g2,  nullptr, c3w,   nullptr, nullptr);
  k_conv<16,16,144,160,16,1,1,2,0><<<cg,128,0,stream>>>(g2, c2w, nullptr, ga1, z1,   nullptr, nullptr, nullptr);
  k_conv<16,8,144,160,8,1,1,3,0><<<cg,128,0,stream>>>(ga1, c1w, nullptr, yf,  nullptr, nullptr, x, dt);

  // X1 aliases ga1 -> zero its K-pad only after F4 consumed ga1.
  k_padX1<<<(NSLICE*HH*30+255)/256, 256, 0, stream>>>(X1);

  k_fft_rows<<<dim3(4,12,512), 128, 0, stream>>>(yf, Mw1t, S1);
  k_cgemm<0><<<dim3(2,12,512), 128, 0, stream>>>(Fr, Fi, S1, (void*)Xf);

  k_zero<<<1, 128, 0, stream>>>(nin, 128);
  k_constrain_norm<<<dim3(73,64), 256, 0, stream>>>(Xf, nin);
  k_scan<<<(BB*CC*HWF+255)/256, 256, 0, stream>>>(Aop, Xf, hf);
  k_constrain_norm<<<dim3(73,64), 256, 0, stream>>>(hf, nout);
  k_finalize<<<(NSLICE*HWF+255)/256, 256, 0, stream>>>(hf, nin, nout, Zt);

  k_cgemm<1><<<dim3(2,12,512), 128, 0, stream>>>(Gr, Gi, Zt, (void*)X1);
  k_ifft_rows<<<dim3(3,12,512), 128, 0, stream>>>(X1, M2t, (float*)d_out);
}
